// AttentionDownsample_29738353557850
// MI455X (gfx1250) — compile-verified, matrix operand order checked by reading, not run on hardware
//
#include <hip/hip_runtime.h>
#include <hip/hip_bf16.h>
#include <cstdint>

// ---------------------------------------------------------------------------
// Problem constants
// ---------------------------------------------------------------------------
#define Bsz      256
#define RES      14
#define NKEYS    196          // RES*RES
#define NQ       49           // ((RES-1)/2+1)^2
#define IN_DIM   384
#define OUT_DIM  512
#define KEY_DIM  16
#define HEADS    12
#define VAL_DIM  32
#define KEY_ATTN 192          // KEY_DIM*HEADS
#define VAL_ATTN 384          // VAL_DIM*HEADS
#define KV_DIM   576          // KEY_ATTN+VAL_ATTN
#define ATT_SCALE 0.25f       // KEY_DIM^-0.5
#define BN_EPS   1e-5f

typedef __attribute__((ext_vector_type(16))) _Float16 v16h;
typedef __attribute__((ext_vector_type(8)))  _Float16 v8h;
typedef __attribute__((ext_vector_type(8)))  float    v8f;

// ---------------------------------------------------------------------------
// f32 -> f16 conversion (x and weight matrices)
// ---------------------------------------------------------------------------
__global__ void cvt_f32_to_f16(const float* __restrict__ src,
                               _Float16* __restrict__ dst, int n) {
  int i = blockIdx.x * blockDim.x + threadIdx.x;
  if (i < n) dst[i] = (_Float16)src[i];
}

// Gather strided query rows: xq[b, qi, c] = x[b, (2*(qi/7))*14 + 2*(qi%7), c]
__global__ void gather_xq(const float* __restrict__ x,
                          _Float16* __restrict__ xqh) {
  int i = blockIdx.x * blockDim.x + threadIdx.x;          // over 12544*384
  if (i >= Bsz * NQ * IN_DIM) return;
  int c   = i % IN_DIM;
  int row = i / IN_DIM;
  int b   = row / NQ;
  int qi  = row % NQ;
  int n   = ((qi / 7) * 2) * RES + (qi % 7) * 2;
  xqh[i] = (_Float16)x[((size_t)(b * NKEYS + n)) * IN_DIM + c];
}

// ---------------------------------------------------------------------------
// WMMA fragment load: 16x32 f16 tile, ld in halves.
// Lane L: row = L&15 ; halves [hi..hi+7] and [16+hi..16+hi+7], hi = (L>>4)*8
// ---------------------------------------------------------------------------
__device__ __forceinline__ v16h load_frag(const _Float16* p, int ld) {
  int lane = threadIdx.x & 31;
  int r    = lane & 15;
  int hi   = (lane >> 4) << 3;
  const _Float16* q = p + (size_t)r * ld + hi;
  v8h lo  = *(const v8h*)(q);
  v8h hi2 = *(const v8h*)(q + 16);
  v16h f;
#pragma unroll
  for (int i = 0; i < 8; ++i) { f[i] = lo[i]; f[8 + i] = hi2[i]; }
  return f;
}

// ---------------------------------------------------------------------------
// Fused GEMM + folded-BN epilogue, software-pipelined.
//   C[M,NC] = A[M,384] @ W[NC,384]^T ; C = C*s + t  (s,t folded BN per col)
// grid = (M/256, NC/64), block = 256 (8 waves: 4 along M x 2 along N)
// Each wave: 64 rows x 32 cols = 4x2 WMMA tiles, 8 v_wmma per k-step,
// next k-step's 6 fragments prefetched before the WMMA burst.
// out_is_f32 ? write float : write f16
// ---------------------------------------------------------------------------
__global__ void gemm_bn(const _Float16* __restrict__ A,
                        const _Float16* __restrict__ W,
                        const float* __restrict__ g,  const float* __restrict__ bb,
                        const float* __restrict__ mm, const float* __restrict__ vv,
                        _Float16* __restrict__ outh, float* __restrict__ outf,
                        int NC, int out_is_f32) {
  const int K = IN_DIM;                       // 384, compile-time for unrolling
  int w  = threadIdx.x >> 5;
  int wm = w & 3;                             // 0..3  (M direction)
  int wn = w >> 2;                            // 0..1  (N direction)
  int rowBase = blockIdx.x * 256 + wm * 64;
  int colBase = blockIdx.y * 64  + wn * 32;

  const _Float16* ap = A + (size_t)rowBase * K;
  const _Float16* wp = W + (size_t)colBase * K;

  v8f acc[4][2];
#pragma unroll
  for (int i = 0; i < 4; ++i)
#pragma unroll
    for (int j = 0; j < 2; ++j) acc[i][j] = v8f{};

  // Prologue: load k-step 0 fragments
  v16h af[4], bf[2];
#pragma unroll
  for (int i = 0; i < 4; ++i) af[i] = load_frag(ap + (size_t)(16 * i) * K, K);
#pragma unroll
  for (int j = 0; j < 2; ++j) bf[j] = load_frag(wp + (size_t)(16 * j) * K, K);

#pragma unroll
  for (int k0 = 0; k0 < K; k0 += 32) {
    v16h an[4], bn2[2];
    const bool has_next = (k0 + 32 < K);
    if (has_next) {
#pragma unroll
      for (int i = 0; i < 4; ++i)
        an[i] = load_frag(ap + (size_t)(16 * i) * K + (k0 + 32), K);
#pragma unroll
      for (int j = 0; j < 2; ++j)
        bn2[j] = load_frag(wp + (size_t)(16 * j) * K + (k0 + 32), K);
    }
#pragma unroll
    for (int i = 0; i < 4; ++i) {
#pragma unroll
      for (int j = 0; j < 2; ++j) {
        acc[i][j] = __builtin_amdgcn_wmma_f32_16x16x32_f16(
            false, af[i], false, bf[j], (short)0, acc[i][j], false, false);
      }
    }
    if (has_next) {
#pragma unroll
      for (int i = 0; i < 4; ++i) af[i] = an[i];
#pragma unroll
      for (int j = 0; j < 2; ++j) bf[j] = bn2[j];
    }
  }

  // Epilogue: BN fold  y = x*s + t ;  s = g*rsqrt(v+eps), t = b - m*s
  int lane = threadIdx.x & 31;
  int rOff = (lane >> 4) * 8;
#pragma unroll
  for (int j = 0; j < 2; ++j) {
    int   col = colBase + j * 16 + (lane & 15);
    float s   = g[col] * rsqrtf(vv[col] + BN_EPS);
    float t   = bb[col] - mm[col] * s;
#pragma unroll
    for (int i = 0; i < 4; ++i) {
#pragma unroll
      for (int e = 0; e < 8; ++e) {
        int    row = rowBase + i * 16 + rOff + e;
        float  val = acc[i][j][e] * s + t;
        size_t idx = (size_t)row * NC + col;
        if (out_is_f32) outf[idx] = val;
        else            outh[idx] = (_Float16)val;
      }
    }
  }
}

// ---------------------------------------------------------------------------
// Attention core: one block (64 threads) per (batch, head).
// kv layout [B*196][576]: per key n, head h -> k = [h*48 .. h*48+15],
//                                              v = [h*48+16 .. h*48+47]
// q  layout [B*49][192]
// Output (post-SiLU) -> aoh [B*49][384] f16, cols h*32..h*32+31
// ---------------------------------------------------------------------------
__global__ void attn_core(const _Float16* __restrict__ kvh,
                          const _Float16* __restrict__ qh,
                          const float* __restrict__ biases,     // [12][196]
                          const int*   __restrict__ bias_idxs,  // [49][196]
                          _Float16* __restrict__ aoh) {
  int b = blockIdx.x;
  int h = blockIdx.y;

  __shared__ _Float16 skv[NKEYS][48];     // k(16) | v(32) per key
  __shared__ float    sbias[NKEYS];       // bias table for this head
  __shared__ float    slog[NQ][NKEYS];    // per-query logits

  int tid = threadIdx.x;                  // 64 threads

  for (int n = tid; n < NKEYS; n += 64) {
    const _Float16* src = kvh + ((size_t)(b * NKEYS + n)) * KV_DIM + h * 48;
    // 48 halves = 96 B contiguous: copy as 6 x v8h
    const v8h* s8 = (const v8h*)src;
    v8h* d8 = (v8h*)&skv[n][0];
#pragma unroll
    for (int d = 0; d < 6; ++d) d8[d] = s8[d];
    sbias[n] = biases[h * NKEYS + n];
  }
  __syncthreads();

  if (tid < NQ) {
    float qv[KEY_DIM];
    const _Float16* qp = qh + ((size_t)(b * NQ + tid)) * KEY_ATTN + h * KEY_DIM;
#pragma unroll
    for (int d = 0; d < KEY_DIM; ++d) qv[d] = (float)qp[d];

    const int* idxrow = bias_idxs + tid * NKEYS;

    float mx = -1e30f;
    for (int n = 0; n < NKEYS; ++n) {
      float dot = 0.f;
#pragma unroll
      for (int d = 0; d < KEY_DIM; ++d) dot += qv[d] * (float)skv[n][d];
      float l = dot * ATT_SCALE + sbias[idxrow[n]];
      slog[tid][n] = l;
      mx = fmaxf(mx, l);
    }
    float sum = 0.f;
    for (int n = 0; n < NKEYS; ++n) {
      float e = __expf(slog[tid][n] - mx);
      slog[tid][n] = e;
      sum += e;
    }
    float inv = 1.f / sum;

    float o[VAL_DIM];
#pragma unroll
    for (int d = 0; d < VAL_DIM; ++d) o[d] = 0.f;
    for (int n = 0; n < NKEYS; ++n) {
      float wgt = slog[tid][n];
#pragma unroll
      for (int d = 0; d < VAL_DIM; ++d) o[d] += wgt * (float)skv[n][16 + d];
    }

    _Float16* op = aoh + ((size_t)(b * NQ + tid)) * VAL_ATTN + h * VAL_DIM;
#pragma unroll
    for (int d = 0; d < VAL_DIM; ++d) {
      float val = o[d] * inv;
      val = val / (1.f + __expf(-val));     // SiLU
      op[d] = (_Float16)val;
    }
  }
}

// ---------------------------------------------------------------------------
// Host-side launcher
// ---------------------------------------------------------------------------
static inline size_t align256(size_t x) { return (x + 255) & ~(size_t)255; }

extern "C" void kernel_launch(void* const* d_in, const int* in_sizes, int n_in,
                              void* d_out, int out_size, void* d_ws, size_t ws_size,
                              hipStream_t stream) {
  const float* x     = (const float*)d_in[0];
  const float* kv_w  = (const float*)d_in[1];
  const float* kv_g  = (const float*)d_in[2];
  const float* kv_b  = (const float*)d_in[3];
  const float* kv_m  = (const float*)d_in[4];
  const float* kv_v  = (const float*)d_in[5];
  const float* q_w   = (const float*)d_in[6];
  const float* q_g   = (const float*)d_in[7];
  const float* q_b   = (const float*)d_in[8];
  const float* q_m   = (const float*)d_in[9];
  const float* q_v   = (const float*)d_in[10];
  const float* pj_w  = (const float*)d_in[11];
  const float* pj_g  = (const float*)d_in[12];
  const float* pj_b  = (const float*)d_in[13];
  const float* pj_m  = (const float*)d_in[14];
  const float* pj_v  = (const float*)d_in[15];
  const float* biases    = (const float*)d_in[16];
  const int*   bias_idxs = (const int*)d_in[17];
  float* out = (float*)d_out;

  // Workspace carve-out (f16 buffers), all 256-byte aligned
  char* ws = (char*)d_ws;
  size_t off = 0;
  _Float16* xh    = (_Float16*)(ws + off); off = align256(off + (size_t)Bsz*NKEYS*IN_DIM*2);
  _Float16* xqh   = (_Float16*)(ws + off); off = align256(off + (size_t)Bsz*NQ*IN_DIM*2);
  _Float16* kvwh  = (_Float16*)(ws + off); off = align256(off + (size_t)KV_DIM*IN_DIM*2);
  _Float16* qwh   = (_Float16*)(ws + off); off = align256(off + (size_t)KEY_ATTN*IN_DIM*2);
  _Float16* pjwh  = (_Float16*)(ws + off); off = align256(off + (size_t)OUT_DIM*IN_DIM*2);
  _Float16* kvh   = (_Float16*)(ws + off); off = align256(off + (size_t)Bsz*NKEYS*KV_DIM*2);
  _Float16* qhb   = (_Float16*)(ws + off); off = align256(off + (size_t)Bsz*NQ*KEY_ATTN*2);
  _Float16* aoh   = (_Float16*)(ws + off); off = align256(off + (size_t)Bsz*NQ*VAL_ATTN*2);

  // 1) f32 -> f16 conversions
  {
    int n = Bsz * NKEYS * IN_DIM;           // 19,267,584
    cvt_f32_to_f16<<<(n + 255) / 256, 256, 0, stream>>>(x, xh, n);
  }
  {
    int n = KV_DIM * IN_DIM;
    cvt_f32_to_f16<<<(n + 255) / 256, 256, 0, stream>>>(kv_w, kvwh, n);
  }
  {
    int n = KEY_ATTN * IN_DIM;
    cvt_f32_to_f16<<<(n + 255) / 256, 256, 0, stream>>>(q_w, qwh, n);
  }
  {
    int n = OUT_DIM * IN_DIM;
    cvt_f32_to_f16<<<(n + 255) / 256, 256, 0, stream>>>(pj_w, pjwh, n);
  }
  {
    int n = Bsz * NQ * IN_DIM;              // 4,816,896
    gather_xq<<<(n + 255) / 256, 256, 0, stream>>>(x, xqh);
  }

  // 2) kv = BN(x @ kv_w^T)  -> f16 [50176][576]
  gemm_bn<<<dim3((Bsz * NKEYS) / 256, KV_DIM / 64), 256, 0, stream>>>(
      xh, kvwh, kv_g, kv_b, kv_m, kv_v, kvh, nullptr, KV_DIM, 0);

  // 3) q = BN(xq @ q_w^T)   -> f16 [12544][192]
  gemm_bn<<<dim3((Bsz * NQ) / 256, KEY_ATTN / 64), 256, 0, stream>>>(
      xqh, qwh, q_g, q_b, q_m, q_v, qhb, nullptr, KEY_ATTN, 0);

  // 4) attention + softmax + SiLU -> f16 [12544][384]
  attn_core<<<dim3(Bsz, HEADS), 64, 0, stream>>>(kvh, qhb, biases, bias_idxs, aoh);

  // 5) out = BN(silu_out @ pj_w^T) -> f32 [12544][512] (final output)
  gemm_bn<<<dim3((Bsz * NQ) / 256, OUT_DIM / 64), 256, 0, stream>>>(
      aoh, pjwh, pj_g, pj_b, pj_m, pj_v, nullptr, out, OUT_DIM, 1);
}
